// SrgcnSoftmaxHead_11879879541099
// MI455X (gfx1250) — matrix operand order repaired and checked by reading, not compiled, not run on hardware
//
#include <hip/hip_runtime.h>
#include <hip/hip_bf16.h>

typedef __attribute__((ext_vector_type(16))) _Float16 v16h;
typedef __attribute__((ext_vector_type(8)))  float    v8f;

#define N_NODES 100000
#define N_EDGES 1600000
#define E_TOT   (N_EDGES + N_NODES)
#define KDIM    128
#define FDIM    64
#define ALPHA   0.2f

// workspace layout (float units)
#define WS_H    0
#define WS_HL   (WS_H  + N_NODES * FDIM)   // 6,400,000
#define WS_HR   (WS_HL + N_NODES)
#define WS_M    (WS_HR + N_NODES)
#define WS_DEN  (WS_M  + N_NODES)
#define WS_E    (WS_DEN + N_NODES)         // 1,700,000 edge scores / exps

// ---------------------------------------------------------------- init
__global__ __launch_bounds__(256) void k_init(float* __restrict__ out,
                                              float* __restrict__ m,
                                              float* __restrict__ den) {
    int i = blockIdx.x * 256 + threadIdx.x;
    if (i < N_NODES * FDIM) out[i] = 0.0f;
    if (i < N_NODES) { m[i] = -__builtin_huge_valf(); den[i] = 0.0f; }
}

// ---------------------------------------------------------------- h = x @ W (WMMA f16 -> f32)
// W is pre-swizzled in LDS into fragment-major order so each lane's B fragment
// is a single contiguous 32-byte read (ds_load_b128 x2) instead of 16 u16 gathers.
__global__ __launch_bounds__(256) void k_gemm(const float* __restrict__ x,
                                              const float* __restrict__ w,
                                              float* __restrict__ h) {
    // Wl[((ks*4 + t)*32 + lane)*16 + j] = (f16) W[k][n]
    //   k = ks*32 + (j>=8 ? 16 : 0) + (lane>=16 ? 8 : 0) + (j&7)   (B: lanes 0-15 hold K {0..7,16..23}, 16-31 hold K {8..15,24..31})
    //   n = t*16 + (lane & 15)
    __shared__ __align__(32) _Float16 Wl[KDIM * FDIM];   // 16 KB
    int tid = threadIdx.x;
    for (int i = tid; i < KDIM * FDIM; i += 256) {
        int j    = i & 15;
        int lane = (i >> 4) & 31;
        int ft   = i >> 9;                    // 0..15 == ks*4 + t
        int k    = (ft >> 2) * 32 + ((j >> 3) << 4) + ((lane >> 4) << 3) + (j & 7);
        int n    = (ft & 3) * 16 + (lane & 15);
        Wl[i] = (_Float16)w[k * FDIM + n];
    }
    __syncthreads();

    int wave  = tid >> 5;
    int lane  = tid & 31;
    int lhalf = lane >> 4;        // 0: lanes 0-15, 1: lanes 16-31
    int lmod  = lane & 15;
    int m0    = blockIdx.x * 128 + wave * 16;     // 8 waves x 16 rows

    const v16h* Bf = (const v16h*)Wl;             // fragment table, v16h-indexed

    v8f acc[4] = {v8f{}, v8f{}, v8f{}, v8f{}};

    #pragma unroll
    for (int ks = 0; ks < 4; ++ks) {              // K = 128 in steps of 32
        // A fragment: 16x32 f16, lane<16 -> K {0..7,16..23}, lane>=16 -> K {8..15,24..31}
        v16h a;
        int mrow = m0 + lmod; if (mrow >= N_NODES) mrow = N_NODES - 1;
        const float* xp = x + mrow * KDIM + ks * 32 + lhalf * 8;
        #pragma unroll
        for (int j = 0; j < 8; ++j) {
            a[j]     = (_Float16)xp[j];
            a[j + 8] = (_Float16)xp[j + 16];
        }
        #pragma unroll
        for (int t = 0; t < 4; ++t) {             // N = 64 in 4 tiles of 16
            v16h b = Bf[(ks * 4 + t) * 32 + lane];
            acc[t] = __builtin_amdgcn_wmma_f32_16x16x32_f16(
                         false, a, false, b, (short)0, acc[t], false, false);
        }
    }

    // C/D layout: VGPR r holds M = lhalf*8 + r, N = lmod (per 16x16 tile)
    #pragma unroll
    for (int t = 0; t < 4; ++t)
        #pragma unroll
        for (int r = 0; r < 8; ++r) {
            int mrow = m0 + lhalf * 8 + r;
            if (mrow < N_NODES) h[mrow * FDIM + t * 16 + lmod] = acc[t][r];
        }
}

// ---------------------------------------------------------------- hl = h.a_l, hr = h.a_r
__global__ __launch_bounds__(256) void k_hlhr(const float* __restrict__ h,
                                              const float* __restrict__ al,
                                              const float* __restrict__ ar,
                                              float* __restrict__ hl,
                                              float* __restrict__ hr) {
    int node = (blockIdx.x * 256 + threadIdx.x) >> 5;
    int lane = threadIdx.x & 31;
    if (node >= N_NODES) return;
    float v0 = h[node * FDIM + lane];
    float v1 = h[node * FDIM + 32 + lane];
    float sl = v0 * al[lane] + v1 * al[lane + 32];
    float sr = v0 * ar[lane] + v1 * ar[lane + 32];
    #pragma unroll
    for (int o = 16; o > 0; o >>= 1) { sl += __shfl_xor(sl, o); sr += __shfl_xor(sr, o); }
    if (lane == 0) { hl[node] = sl; hr[node] = sr; }
}

// ---------------------------------------------------------------- float atomic max (sign-split)
__device__ __forceinline__ void atomicMaxF(float* a, float v) {
    if (v >= 0.0f) atomicMax((int*)a, __float_as_int(v));
    else           atomicMin((unsigned int*)a, __float_as_uint(v));
}

// ---------------------------------------------------------------- edge pass 1: score + segment max
__global__ __launch_bounds__(256) void k_score(const long long* __restrict__ ei,
                                               const float* __restrict__ hl,
                                               const float* __restrict__ hr,
                                               float* __restrict__ sc,
                                               float* __restrict__ m) {
    int e = blockIdx.x * 256 + threadIdx.x;
    if (e >= E_TOT) return;
    int r, c;
    if (e < N_EDGES) { r = (int)ei[e]; c = (int)ei[N_EDGES + e]; }
    else             { r = c = e - N_EDGES; }
    float s = hl[r] + hr[c];
    s = s > 0.0f ? s : ALPHA * s;                 // leaky_relu
    sc[e] = s;
    atomicMaxF(m + r, s);
}

// ---------------------------------------------------------------- edge pass 2: exp + segment sum
__global__ __launch_bounds__(256) void k_expsum(const long long* __restrict__ ei,
                                                const float* __restrict__ m,
                                                float* __restrict__ sc,
                                                float* __restrict__ den) {
    int e = blockIdx.x * 256 + threadIdx.x;
    if (e >= E_TOT) return;
    int r = (e < N_EDGES) ? (int)ei[e] : (e - N_EDGES);
    float v = __expf(sc[e] - m[r]);
    sc[e] = v;
    atomicAdd(den + r, v);
}

// ---------------------------------------------------------------- edge pass 3: SpMM scatter
__global__ __launch_bounds__(256) void k_spmm(const long long* __restrict__ ei,
                                              const float* __restrict__ sc,
                                              const float* __restrict__ den,
                                              const float* __restrict__ h,
                                              float* __restrict__ out) {
    int e = (blockIdx.x * 256 + threadIdx.x) >> 5;   // one wave32 per edge
    int lane = threadIdx.x & 31;
    if (e >= E_TOT) return;
    int r, c;
    if (e < N_EDGES) { r = (int)ei[e]; c = (int)ei[N_EDGES + e]; }
    else             { r = c = e - N_EDGES; }
    float wgt = sc[e] / den[r];
    float v0 = h[c * FDIM + lane];
    float v1 = h[c * FDIM + 32 + lane];
    atomicAdd(out + r * FDIM + lane,      wgt * v0);
    atomicAdd(out + r * FDIM + 32 + lane, wgt * v1);
}

// ---------------------------------------------------------------- bias + nan fix + adaptive gate
__global__ __launch_bounds__(256) void k_final(const float* __restrict__ bias,
                                               const float* __restrict__ fc,
                                               const float* __restrict__ bf,
                                               float* __restrict__ out) {
    int node = (blockIdx.x * 256 + threadIdx.x) >> 5;
    int lane = threadIdx.x & 31;
    if (node >= N_NODES) return;
    float v0 = out[node * FDIM + lane];
    float v1 = out[node * FDIM + 32 + lane];
    // nan -> 0, +/-inf -> +/-FLT_MAX (jnp.nan_to_num defaults)
    v0 = (v0 == v0) ? fminf(fmaxf(v0, -3.4028235e38f), 3.4028235e38f) : 0.0f;
    v1 = (v1 == v1) ? fminf(fmaxf(v1, -3.4028235e38f), 3.4028235e38f) : 0.0f;
    v0 += bias[lane];
    v1 += bias[lane + 32];
    float s = v0 * fc[lane] + v1 * fc[lane + 32];
    #pragma unroll
    for (int o = 16; o > 0; o >>= 1) s += __shfl_xor(s, o);
    float g = 1.0f / (1.0f + __expf(-(s + bf[0])));
    out[node * FDIM + lane]      = fmaxf(v0, 0.0f) + g * fminf(v0, 0.0f);
    out[node * FDIM + 32 + lane] = fmaxf(v1, 0.0f) + g * fminf(v1, 0.0f);
}

// ---------------------------------------------------------------- launch
extern "C" void kernel_launch(void* const* d_in, const int* in_sizes, int n_in,
                              void* d_out, int out_size, void* d_ws, size_t ws_size,
                              hipStream_t stream) {
    (void)in_sizes; (void)n_in; (void)out_size; (void)ws_size;
    const float*     x    = (const float*)d_in[0];
    const long long* ei   = (const long long*)d_in[1];
    const float*     w    = (const float*)d_in[2];
    const float*     bias = (const float*)d_in[3];
    const float*     al   = (const float*)d_in[4];
    const float*     ar   = (const float*)d_in[5];
    const float*     fc   = (const float*)d_in[6];
    const float*     bf   = (const float*)d_in[7];
    float* out = (float*)d_out;
    float* wsf = (float*)d_ws;
    float* h   = wsf + WS_H;
    float* hl  = wsf + WS_HL;
    float* hr  = wsf + WS_HR;
    float* m   = wsf + WS_M;
    float* den = wsf + WS_DEN;
    float* sc  = wsf + WS_E;

    k_init  <<<(N_NODES * FDIM + 255) / 256, 256, 0, stream>>>(out, m, den);
    k_gemm  <<<(N_NODES + 127) / 128,        256, 0, stream>>>(x, w, h);
    k_hlhr  <<<(N_NODES + 7) / 8,            256, 0, stream>>>(h, al, ar, hl, hr);
    k_score <<<(E_TOT + 255) / 256,          256, 0, stream>>>(ei, hl, hr, sc, m);
    k_expsum<<<(E_TOT + 255) / 256,          256, 0, stream>>>(ei, m, sc, den);
    k_spmm  <<<(E_TOT + 7) / 8,              256, 0, stream>>>(ei, sc, den, h, out);
    k_final <<<(N_NODES + 7) / 8,            256, 0, stream>>>(bias, fc, bf, out);
}
